// MaxPoolingAttn_13159779795000
// MI455X (gfx1250) — matrix operand order repaired and checked by reading, not compiled, not run on hardware
//
#include <hip/hip_runtime.h>
#include <hip/hip_bf16.h>

typedef __bf16 bf16_t;
typedef __attribute__((ext_vector_type(16))) __bf16 v16bf;
typedef __attribute__((ext_vector_type(8)))  __bf16 v8bf;
typedef __attribute__((ext_vector_type(8)))  float  v8f;

#define HW 4096   // 64*64 spatial
#define IMW 64

__device__ __forceinline__ float ssslu_f(float x, float shape, float bias) {
    float s = x / (1.0f + fabsf(x));
    return x * (__sinf(s * (shape * 1.5707963267948966f)) + (bias + 1.0f));
}
__device__ __forceinline__ float sigmoid_f(float x) { return 1.0f / (1.0f + __expf(-x)); }
__device__ __forceinline__ float bn_f(float x, float g, float b, float m, float v) {
    return (x - m) * (g * rsqrtf(v + 1e-3f)) + b;
}

// ---------------- weight prep ----------------
__global__ void f32_to_bf16_kernel(const float* __restrict__ in, bf16_t* __restrict__ out, int n) {
    int i = blockIdx.x * 256 + threadIdx.x;
    if (i < n) out[i] = (bf16_t)in[i];
}

// [O][I][3][3] f32 -> [O][9][I] bf16 (K = tap*I + c contiguous for WMMA A-rows)
__global__ void reorder3x3_kernel(const float* __restrict__ in, bf16_t* __restrict__ out, int O, int I) {
    int i = blockIdx.x * 256 + threadIdx.x;
    int n = O * I * 9;
    if (i >= n) return;
    int o = i / (I * 9);
    int rem = i % (I * 9);
    int tap = rem / I;
    int c = rem % I;
    out[i] = (bf16_t)in[((size_t)o * I + c) * 9 + tap];
}

// ---------------- 1x1 conv as GEMM (WMMA bf16, software-pipelined) ----------------
// MODE 0: qkv  (K=256, M=768, input f32)  epilogue: ssslu(-3,.3) -> BN;
//         c<512 -> t_f32, c>=512 -> concat bf16 (channel c-256)
// MODE 3: back2 (K=256, M=256, input bf16) epilogue: +bias -> BN -> ssslu(.8,0) -> sigmoid -> f32 out
template <int MODE>
__global__ __launch_bounds__(128)
void gemm1x1_wmma(const bf16_t* __restrict__ A,
                  const float* __restrict__ Xf, const bf16_t* __restrict__ Xb,
                  int K,
                  const float* __restrict__ bn_g, const float* __restrict__ bn_b,
                  const float* __restrict__ bn_m, const float* __restrict__ bn_v,
                  const float* __restrict__ bias,
                  float* __restrict__ outF, bf16_t* __restrict__ outB) {
    constexpr int LDB = 48;                       // padded row (96B, 32B-aligned frags)
    __shared__ __attribute__((aligned(32))) bf16_t ldsB[64 * LDB];
    const int tid = threadIdx.x;
    const int wave = tid >> 5;
    const int laneLo = tid & 15;
    const int laneHi = (tid >> 4) & 1;
    const int b = blockIdx.z;
    const int mBase = blockIdx.x * 64 + wave * 16;
    const int nBase = blockIdx.y * 64;
    const float*  xfB = (MODE == 0) ? (Xf + (size_t)b * K * HW + nBase) : nullptr;
    const bf16_t* xbB = (MODE == 3) ? (Xb + (size_t)b * K * HW + nBase) : nullptr;
    const bf16_t* Arow0 = A + (size_t)(mBase + laneLo) * K + laneHi * 16;

    v8f acc[4];
#pragma unroll
    for (int t = 0; t < 4; ++t)
#pragma unroll
        for (int r = 0; r < 8; ++r) acc[t][r] = 0.0f;

    // register staging buffers (whole 32x64 tile share per thread, vector loads)
    float4 regF[4];   // MODE 0: 4 x float4 = 16 f32
    v8bf   regB[2];   // MODE 3: 2 x 8 bf16 = 16 bf16

    auto loadTile = [&](int k0) {
        if (MODE == 0) {
#pragma unroll
            for (int it = 0; it < 4; ++it) {
                int idx = it * 512 + tid * 4;        // 2048 elems / 128 thr / 4-wide
                int kk = idx >> 6, n0 = idx & 63;
                regF[it] = *(const float4*)&xfB[(k0 + kk) * HW + n0];
            }
        } else {
#pragma unroll
            for (int it = 0; it < 2; ++it) {
                int idx = it * 1024 + tid * 8;       // 8-wide bf16 loads
                int kk = idx >> 6, n0 = idx & 63;
                regB[it] = *(const v8bf*)&xbB[(k0 + kk) * HW + n0];
            }
        }
    };
    auto storeTile = [&]() {
        if (MODE == 0) {
#pragma unroll
            for (int it = 0; it < 4; ++it) {
                int idx = it * 512 + tid * 4;
                int kk = idx >> 6, n0 = idx & 63;
                const float* f = &regF[it].x;
#pragma unroll
                for (int j = 0; j < 4; ++j) ldsB[(n0 + j) * LDB + kk] = (bf16_t)f[j];
            }
        } else {
#pragma unroll
            for (int it = 0; it < 2; ++it) {
                int idx = it * 1024 + tid * 8;
                int kk = idx >> 6, n0 = idx & 63;
#pragma unroll
                for (int j = 0; j < 8; ++j) ldsB[(n0 + j) * LDB + kk] = regB[it][j];
            }
        }
    };

    loadTile(0);
    for (int k0 = 0; k0 < K; k0 += 32) {
        storeTile();
        __syncthreads();
        if (k0 + 32 < K) loadTile(k0 + 32);          // overlap next tile with WMMA below
        const bf16_t* Arow = Arow0 + k0;
        v8bf a0 = *(const v8bf*)(Arow);
        v8bf a1 = *(const v8bf*)(Arow + 8);
        v16bf af;
#pragma unroll
        for (int i = 0; i < 8; ++i) { af[i] = a0[i]; af[i + 8] = a1[i]; }
#pragma unroll
        for (int t = 0; t < 4; ++t) {
            v16bf bfrag = *(const v16bf*)&ldsB[(t * 16 + laneLo) * LDB + laneHi * 16];
            acc[t] = __builtin_amdgcn_wmma_f32_16x16x32_bf16(false, af, false, bfrag,
                                                             (short)0, acc[t], false, false);
        }
        __syncthreads();
    }

#pragma unroll
    for (int r = 0; r < 8; ++r) {                 // D layout: m = r + 8*laneHi, n = laneLo
        int mm = mBase + r + 8 * laneHi;
        float g = bn_g[mm], bb = bn_b[mm], mn = bn_m[mm], vr = bn_v[mm];
        float bi = (MODE == 3) ? bias[mm] : 0.0f;
#pragma unroll
        for (int t = 0; t < 4; ++t) {
            int nn = nBase + t * 16 + laneLo;
            float z = acc[t][r];
            if (MODE == 0) {
                z = bn_f(ssslu_f(z, -3.0f, 0.3f), g, bb, mn, vr);
                if (mm < 512) outF[((size_t)(b * 512 + mm)) * HW + nn] = z;
                else          outB[((size_t)(b * 512 + (mm - 256))) * HW + nn] = (bf16_t)z;
            } else {
                z = ssslu_f(bn_f(z + bi, g, bb, mn, vr), 0.8f, 0.0f);
                outF[((size_t)(b * 256 + mm)) * HW + nn] = sigmoid_f(z);
            }
        }
    }
}

// ---------------- 3x3 conv as implicit GEMM (WMMA bf16, software-pipelined) ----------------
// MODE 1: co   epilogue: BN -> ssslu(1,0)           -> bf16
// MODE 2: b1   epilogue: +bias -> ssslu(-3,.3) -> BN -> bf16
template <int MODE>
__global__ __launch_bounds__(128)
void conv3x3_wmma(const bf16_t* __restrict__ A,   // [256][9][Cin]
                  const bf16_t* __restrict__ X,   // [B][Cin][64][64]
                  int Cin,
                  const float* __restrict__ bn_g, const float* __restrict__ bn_b,
                  const float* __restrict__ bn_m, const float* __restrict__ bn_v,
                  const float* __restrict__ bias,
                  bf16_t* __restrict__ out) {
    constexpr int LDB = 48;
    __shared__ __attribute__((aligned(32))) bf16_t ldsB[64 * LDB];
    const int tid = threadIdx.x;
    const int wave = tid >> 5;
    const int laneLo = tid & 15;
    const int laneHi = (tid >> 4) & 1;
    const int b = blockIdx.z;
    const int h = blockIdx.y;                     // output row; N tile = one image row
    const int mBase = blockIdx.x * 64 + wave * 16;
    const int Ktot = 9 * Cin;
    const int CS = Cin >> 5;                      // chunks of 32 channels
    const int S = 9 * CS;                         // total K steps
    const bf16_t* Xb = X + (size_t)b * Cin * HW;
    const bf16_t* Arow0 = A + (size_t)(mBase + laneLo) * Ktot + laneHi * 16;

    v8f acc[4];
#pragma unroll
    for (int t = 0; t < 4; ++t)
#pragma unroll
        for (int r = 0; r < 8; ++r) acc[t][r] = 0.0f;

    v8bf regB[2];                                 // 16 bf16 per thread per tile

    auto loadTile = [&](int s) {
        int tap = s / CS, c0 = (s - tap * CS) * 32;
        int dy = tap / 3 - 1;
        int hy = h + dy;
        bool rowOK = (hy >= 0) && (hy < IMW);
#pragma unroll
        for (int it = 0; it < 2; ++it) {
            int idx = it * 1024 + tid * 8;
            int kk = idx >> 6, u0 = idx & 63;     // aligned source columns u0..u0+7
            if (rowOK) regB[it] = *(const v8bf*)&Xb[(c0 + kk) * HW + hy * IMW + u0];
            else {
#pragma unroll
                for (int j = 0; j < 8; ++j) regB[it][j] = (bf16_t)0.0f;
            }
        }
    };
    auto storeTile = [&](int s) {
        int tap = s / CS;
        int dx = tap % 3 - 1;
        // lds[w][kk] = src[u = w+dx]  ->  loaded u goes to row w = u - dx
        if (dx != 0 && tid < 32) ldsB[(dx == 1 ? 63 : 0) * LDB + tid] = (bf16_t)0.0f;
#pragma unroll
        for (int it = 0; it < 2; ++it) {
            int idx = it * 1024 + tid * 8;
            int kk = idx >> 6, u0 = idx & 63;
#pragma unroll
            for (int j = 0; j < 8; ++j) {
                int w = u0 + j - dx;
                if (w >= 0 && w < IMW) ldsB[w * LDB + kk] = regB[it][j];
            }
        }
    };

    loadTile(0);
    for (int s = 0; s < S; ++s) {
        storeTile(s);
        __syncthreads();
        if (s + 1 < S) loadTile(s + 1);           // overlap next tile with WMMA below
        int tap = s / CS, c0 = (s - tap * CS) * 32;
        const bf16_t* Arow = Arow0 + tap * Cin + c0;
        v8bf a0 = *(const v8bf*)(Arow);
        v8bf a1 = *(const v8bf*)(Arow + 8);
        v16bf af;
#pragma unroll
        for (int i = 0; i < 8; ++i) { af[i] = a0[i]; af[i + 8] = a1[i]; }
#pragma unroll
        for (int t = 0; t < 4; ++t) {
            v16bf bfrag = *(const v16bf*)&ldsB[(t * 16 + laneLo) * LDB + laneHi * 16];
            acc[t] = __builtin_amdgcn_wmma_f32_16x16x32_bf16(false, af, false, bfrag,
                                                             (short)0, acc[t], false, false);
        }
        __syncthreads();
    }

#pragma unroll
    for (int r = 0; r < 8; ++r) {
        int mm = mBase + r + 8 * laneHi;
        float g = bn_g[mm], bb = bn_b[mm], mn = bn_m[mm], vr = bn_v[mm];
        float bi = (MODE == 2) ? bias[mm] : 0.0f;
#pragma unroll
        for (int t = 0; t < 4; ++t) {
            int w = t * 16 + laneLo;
            float z = acc[t][r];
            if (MODE == 1) z = ssslu_f(bn_f(z, g, bb, mn, vr), 1.0f, 0.0f);
            else           z = bn_f(ssslu_f(z + bi, -3.0f, 0.3f), g, bb, mn, vr);
            out[((size_t)(b * 256 + mm)) * HW + h * IMW + w] = (bf16_t)z;
        }
    }
}

// ---------------- pooled attention (algebraically reduced) ----------------
// One 64-thread block per (b,q). max_a qw[a]*kw[b] == kw[b]*(kw[b]>=0 ? max(qw) : min(qw)).
__global__ __launch_bounds__(64)
void attn_pool_kernel(const float* __restrict__ t, bf16_t* __restrict__ concat) {
    const int bi = blockIdx.x;
    const int b = bi >> 8, q = bi & 255;
    const float* xq = t + ((size_t)(b * 512 + q)) * HW;
    const float* xk = t + ((size_t)(b * 512 + 256 + q)) * HW;
    __shared__ float tile[IMW * IMW];
    __shared__ float qw[IMW], kw[IMW], qh[IMW], kh[IMW];
    __shared__ float xaw[IMW], xah[IMW];
    const int tid = threadIdx.x;
    const int col0 = (tid & 15) * 4;              // float4 column group
    const int rsub = tid >> 4;                    // row-within-group

    auto loadImg = [&](const float* src) {
#pragma unroll
        for (int pass = 0; pass < 4; ++pass) {
            float4 v[4];
#pragma unroll
            for (int i = 0; i < 4; ++i) {         // 4 outstanding vector loads
                int row = (pass * 4 + i) * 4 + rsub;
                v[i] = *(const float4*)&src[row * IMW + col0];
            }
#pragma unroll
            for (int i = 0; i < 4; ++i) {
                int row = (pass * 4 + i) * 4 + rsub;
                *(float4*)&tile[row * IMW + col0] = v[i];
            }
        }
    };

    loadImg(xq);
    __syncthreads();
    {
        float cm = -3.4e38f, rm = -3.4e38f;
        for (int i = 0; i < IMW; ++i) {
            cm = fmaxf(cm, tile[i * IMW + tid]);   // max over h  -> qw[w]
            rm = fmaxf(rm, tile[tid * IMW + i]);   // max over w  -> qh[h]
        }
        qw[tid] = cm; qh[tid] = rm;
    }
    __syncthreads();
    loadImg(xk);
    __syncthreads();
    {
        float cm = -3.4e38f, rm = -3.4e38f;
        for (int i = 0; i < IMW; ++i) {
            cm = fmaxf(cm, tile[i * IMW + tid]);
            rm = fmaxf(rm, tile[tid * IMW + i]);
        }
        kw[tid] = cm; kh[tid] = rm;
    }
    __syncthreads();

    float qwmax = -3.4e38f, qwmin = 3.4e38f, khmax = -3.4e38f, khmin = 3.4e38f;
    for (int i = 0; i < IMW; ++i) {
        qwmax = fmaxf(qwmax, qw[i]); qwmin = fminf(qwmin, qw[i]);
        khmax = fmaxf(khmax, kh[i]); khmin = fminf(khmin, kh[i]);
    }
    float aw = kw[tid]; aw = (aw >= 0.0f) ? aw * qwmax : aw * qwmin;
    float ah = qh[tid]; ah = (ah >= 0.0f) ? ah * khmax : ah * khmin;
    xaw[tid] = aw; xah[tid] = ah;
    __syncthreads();

    float sw = 0.f, sw2 = 0.f, sh = 0.f, sh2 = 0.f;
    for (int i = 0; i < IMW; ++i) {
        float u = xaw[i]; sw += u; sw2 += u * u;
        float p = xah[i]; sh += p; sh2 += p * p;
    }
    float muw = sw * (1.0f / IMW), varw = sw2 * (1.0f / IMW) - muw * muw;
    float muh = sh * (1.0f / IMW), varh = sh2 * (1.0f / IMW) - muh * muh;
    float hw_ = (aw - muw) * rsqrtf(varw + 1e-5f);
    float hh_ = (ah - muh) * rsqrtf(varh + 1e-5f);
    float aw2 = hw_ * sigmoid_f(hw_);
    float ah2 = hh_ * sigmoid_f(hh_);
    __syncthreads();
    xaw[tid] = aw2; xah[tid] = ah2;
    __syncthreads();

    bf16_t* dst = concat + ((size_t)(b * 512 + q)) * HW;   // xa occupies channels 0..255
    float myw = xaw[tid];
    for (int hh = 0; hh < IMW; ++hh)
        dst[hh * IMW + tid] = (bf16_t)(xah[hh] * myw);
}

// ---------------- launcher ----------------
extern "C" void kernel_launch(void* const* d_in, const int* in_sizes, int n_in,
                              void* d_out, int out_size, void* d_ws, size_t ws_size,
                              hipStream_t stream) {
    (void)in_sizes; (void)n_in; (void)out_size; (void)ws_size;
    const float* x       = (const float*)d_in[0];
    const float* w_qkv   = (const float*)d_in[1];
    const float* g_qkv   = (const float*)d_in[2];
    const float* b_qkv   = (const float*)d_in[3];
    const float* m_qkv   = (const float*)d_in[4];
    const float* v_qkv   = (const float*)d_in[5];
    const float* w_co    = (const float*)d_in[6];
    const float* g_co    = (const float*)d_in[7];
    const float* b_co    = (const float*)d_in[8];
    const float* m_co    = (const float*)d_in[9];
    const float* v_co    = (const float*)d_in[10];
    const float* w_b1    = (const float*)d_in[11];
    const float* bias_b1 = (const float*)d_in[12];
    const float* g_b1    = (const float*)d_in[13];
    const float* b_b1    = (const float*)d_in[14];
    const float* m_b1    = (const float*)d_in[15];
    const float* v_b1    = (const float*)d_in[16];
    const float* w_b2    = (const float*)d_in[17];
    const float* bias_b2 = (const float*)d_in[18];
    const float* g_b2    = (const float*)d_in[19];
    const float* b_b2    = (const float*)d_in[20];
    const float* m_b2    = (const float*)d_in[21];
    const float* v_b2    = (const float*)d_in[22];
    float* out = (float*)d_out;

    char* wsc = (char*)d_ws;
    size_t off = 0;
    auto take = [&](size_t bytes) -> char* {
        char* p = wsc + off;
        off += (bytes + 255) & ~(size_t)255;
        return p;
    };
    float*  t_f32  = (float*) take((size_t)16 * 512 * HW * 4);   // q,k (post SSSLU+BN)
    bf16_t* concat = (bf16_t*)take((size_t)16 * 512 * HW * 2);   // [xa | xv] bf16
    bf16_t* y1     = (bf16_t*)take((size_t)16 * 256 * HW * 2);   // co output
    bf16_t* y2     = (bf16_t*)take((size_t)16 * 256 * HW * 2);   // back1 output
    bf16_t* wqkvB  = (bf16_t*)take((size_t)768 * 256 * 2);
    bf16_t* wcoR   = (bf16_t*)take((size_t)256 * 9 * 512 * 2);
    bf16_t* wb1R   = (bf16_t*)take((size_t)256 * 9 * 256 * 2);
    bf16_t* wb2B   = (bf16_t*)take((size_t)256 * 256 * 2);

    // weight prep
    f32_to_bf16_kernel<<<(768 * 256 + 255) / 256, 256, 0, stream>>>(w_qkv, wqkvB, 768 * 256);
    reorder3x3_kernel<<<(256 * 512 * 9 + 255) / 256, 256, 0, stream>>>(w_co, wcoR, 256, 512);
    reorder3x3_kernel<<<(256 * 256 * 9 + 255) / 256, 256, 0, stream>>>(w_b1, wb1R, 256, 256);
    f32_to_bf16_kernel<<<(256 * 256 + 255) / 256, 256, 0, stream>>>(w_b2, wb2B, 256 * 256);

    // qkv conv1x1 + SSSLU(-3,.3) + BN  ->  t (q,k) f32 + xv bf16 in concat
    gemm1x1_wmma<0><<<dim3(12, 64, 16), 128, 0, stream>>>(
        wqkvB, x, nullptr, 256, g_qkv, b_qkv, m_qkv, v_qkv, nullptr, t_f32, concat);

    // pooled attention -> xa bf16 into concat channels 0..255
    attn_pool_kernel<<<4096, 64, 0, stream>>>(t_f32, concat);

    // co conv3x3 (512->256) + BN + SSSLU(1,0) -> y1 bf16
    conv3x3_wmma<1><<<dim3(4, 64, 16), 128, 0, stream>>>(
        wcoR, concat, 512, g_co, b_co, m_co, v_co, nullptr, y1);

    // back1 conv3x3 (256->256,+bias) + SSSLU(-3,.3) + BN -> y2 bf16
    conv3x3_wmma<2><<<dim3(4, 64, 16), 128, 0, stream>>>(
        wb1R, y1, 256, g_b1, b_b1, m_b1, v_b1, bias_b1, y2);

    // back2 conv1x1 (+bias) + BN + SSSLU(.8,0) + sigmoid -> f32 out
    gemm1x1_wmma<3><<<dim3(4, 64, 16), 128, 0, stream>>>(
        wb2B, nullptr, y2, 256, g_b2, b_b2, m_b2, v_b2, bias_b2, out, nullptr);
}
